// GroupedQueryAttention_48326972015310
// MI455X (gfx1250) — matrix operand order repaired and checked by reading, not compiled, not run on hardware
//
#include <hip/hip_runtime.h>

#define B_ 4
#define T_ 2048
#define C_ 2048
#define H_ 16
#define KV_ 4
#define D_ 128
#define G_ (H_ / KV_)

typedef __attribute__((ext_vector_type(16))) __bf16 v16bf;
typedef __attribute__((ext_vector_type(8)))  __bf16 v8bf;
typedef __attribute__((ext_vector_type(8)))  float  v8f;

// ---------- helpers ----------
__device__ __forceinline__ unsigned short f2bf(float f) {
    union { float f; unsigned u; } v; v.f = f;
    unsigned r = v.u + 0x7FFFu + ((v.u >> 16) & 1u);   // RNE
    return (unsigned short)(r >> 16);
}
__device__ __forceinline__ float bf2f(unsigned short u) {
    union { unsigned u; float f; } v; v.u = ((unsigned)u) << 16;
    return v.f;
}
// A-matrix 16x32 bf16 fragment: per-lane two contiguous 8-elem (16B) chunks.
__device__ __forceinline__ v16bf ldfragA(const unsigned short* p0, const unsigned short* p1) {
    union { v16bf v; v8bf h[2]; } u;
    u.h[0] = *(const v8bf*)p0;
    u.h[1] = *(const v8bf*)p1;
    return u.v;
}
__device__ __forceinline__ v8f wmma_bf16(v16bf a, v16bf b, v8f c) {
    return __builtin_amdgcn_wmma_f32_16x16x32_bf16(false, a, false, b, (short)0, c, false, false);
}
// gfx1250 async global->LDS copy (16B per lane), tracked by ASYNCcnt.
// lds_off = LDS byte offset (low 32b of generic shared pointer), gaddr = 64-bit global addr.
__device__ __forceinline__ void async_ld_b128(unsigned lds_off, unsigned long long gaddr) {
    asm volatile("global_load_async_to_lds_b128 %0, %1, off"
                 :: "v"(lds_off), "v"(gaddr) : "memory");
}
__device__ __forceinline__ void wait_asynccnt0() {
    asm volatile("s_wait_asynccnt 0x0" ::: "memory");
}

// ---------- stage 0: conversions ----------
__global__ __launch_bounds__(256) void cvt_f32_bf16(const float* __restrict__ in,
                                                    unsigned short* __restrict__ out, int n) {
    int i = blockIdx.x * 256 + threadIdx.x;
    if (i < n) out[i] = f2bf(in[i]);
}

__global__ __launch_bounds__(256) void pack_wqkv(const float* __restrict__ Wq,
                                                 const float* __restrict__ Wk,
                                                 const float* __restrict__ Wv,
                                                 unsigned short* __restrict__ out) {
    int i = blockIdx.x * 256 + threadIdx.x;          // over 2048*3072
    if (i >= C_ * 3072) return;
    int k = i / 3072, n = i % 3072;
    float v;
    if (n < 2048)      v = Wq[k * 2048 + n];
    else if (n < 2560) v = Wk[k * 512 + (n - 2048)];
    else               v = Wv[k * 512 + (n - 2560)];
    out[i] = f2bf(v);
}

// ---------- tiled bf16 WMMA GEMM: C[M,N] = A[M,K] * W[K,N] ----------
template<bool OUT_BF16>
__global__ __launch_bounds__(256) void gemm_bf16(const unsigned short* __restrict__ A,
                                                 const unsigned short* __restrict__ W,
                                                 void* __restrict__ Cout,
                                                 int M, int N, int K) {
    __shared__ unsigned short As[128 * 48];   // [row][k] stride 48 (96B, 16B aligned)
    __shared__ unsigned short Bs[128 * 48];   // [n][k] N-major, stride 48

    const int m0 = blockIdx.y * 128;
    const int n0 = blockIdx.x * 128;
    const int tid  = threadIdx.x;
    const int wid  = tid >> 5, lane = tid & 31;
    const int h    = lane >> 4, ln = lane & 15;
    const int wm0  = (wid >> 2) * 64;   // 2 wave rows
    const int wn0  = (wid & 3) * 32;    // 4 wave cols

    v8f acc[4][2];
    #pragma unroll
    for (int im = 0; im < 4; im++)
        #pragma unroll
        for (int in = 0; in < 2; in++)
            acc[im][in] = (v8f){0.f,0.f,0.f,0.f,0.f,0.f,0.f,0.f};

    for (int k0 = 0; k0 < K; k0 += 32) {
        // A tile 128x32 -> LDS via async global->LDS (no rearrangement needed)
        {
            int row = tid >> 1, kk = (tid & 1) * 16;
            const unsigned short* gsrc = A + (size_t)(m0 + row) * K + k0 + kk;
            unsigned la = (unsigned)(size_t)&As[row * 48 + kk];
            async_ld_b128(la,      (unsigned long long)(size_t)gsrc);
            async_ld_b128(la + 16, (unsigned long long)(size_t)gsrc + 16);
            if (k0 + 32 < K) __builtin_prefetch(gsrc + 32, 0, 1);
        }
        // W tile 32x128 -> LDS transposed (8 threads per k row, 16 n each)
        {
            int kr = tid >> 3, n = (tid & 7) * 16;
            const uint4* s = (const uint4*)(W + (size_t)(k0 + kr) * N + n0 + n);
            uint4 tmp[2]; tmp[0] = s[0]; tmp[1] = s[1];
            const unsigned short* w = (const unsigned short*)tmp;
            #pragma unroll
            for (int j = 0; j < 16; j++) Bs[(n + j) * 48 + kr] = w[j];
            if (k0 + 32 < K) __builtin_prefetch(W + (size_t)(k0 + 32 + kr) * N + n0 + n, 0, 1);
        }
        wait_asynccnt0();
        __syncthreads();

        v16bf af[4], bf[2];
        #pragma unroll
        for (int im = 0; im < 4; im++) {
            int m = wm0 + im * 16 + ln;
            af[im] = ldfragA(&As[m * 48 + h * 16], &As[m * 48 + h * 16 + 8]);
        }
        #pragma unroll
        for (int in = 0; in < 2; in++) {
            int n = wn0 + in * 16 + ln;
            bf[in] = *(const v16bf*)&Bs[n * 48 + h * 16];   // contiguous 16 K elems per lane
        }
        #pragma unroll
        for (int im = 0; im < 4; im++)
            #pragma unroll
            for (int in = 0; in < 2; in++)
                acc[im][in] = wmma_bf16(af[im], bf[in], acc[im][in]);
        __syncthreads();
    }

    // epilogue: C layout lane(n=ln, half h) holds rows r+8h
    #pragma unroll
    for (int im = 0; im < 4; im++)
        #pragma unroll
        for (int in = 0; in < 2; in++)
            #pragma unroll
            for (int r = 0; r < 8; r++) {
                int m = m0 + wm0 + im * 16 + r + 8 * h;
                int n = n0 + wn0 + in * 16 + ln;
                float v = acc[im][in][r];
                if (OUT_BF16) ((unsigned short*)Cout)[(size_t)m * N + n] = f2bf(v);
                else          ((float*)Cout)[(size_t)m * N + n] = v;
            }
}

// ---------- RoPE + rearrange to head-major ----------
__global__ __launch_bounds__(256) void rope_rearrange(const unsigned short* __restrict__ qkv,
                                                      unsigned short* __restrict__ qr,
                                                      unsigned short* __restrict__ kr,
                                                      unsigned short* __restrict__ vr) {
    const int bt = blockIdx.x;
    const int b = bt / T_, t = bt % T_;
    const unsigned short* row = qkv + (size_t)bt * 3072;
    const float LOG_BASE_OVER_HALF = 0.14391156f; // ln(10000)/64

    for (int c = threadIdx.x; c < 3072; c += 256) {
        if (c < 2048) {                                     // Q with RoPE
            int hq = c >> 7, d = c & 127;
            int dp = (d < 64) ? d + 64 : d - 64;
            float val = bf2f(row[c]);
            float oth = bf2f(row[(c & ~127) + dp]);
            float fi  = (float)(d & 63);
            float ang = (float)t * __expf(-LOG_BASE_OVER_HALF * fi);
            float cs = __cosf(ang), sn = __sinf(ang);
            float o = (d < 64) ? (val * cs - oth * sn) : (val * cs + oth * sn);
            qr[(((size_t)b * H_ + hq) * T_ + t) * D_ + d] = f2bf(o);
        } else if (c < 2560) {                              // K with RoPE
            int idx = c - 2048;
            int kv = idx >> 7, d = idx & 127;
            int dp = (d < 64) ? d + 64 : d - 64;
            float val = bf2f(row[c]);
            float oth = bf2f(row[(c & ~127) + dp]);
            float fi  = (float)(d & 63);
            float ang = (float)t * __expf(-LOG_BASE_OVER_HALF * fi);
            float cs = __cosf(ang), sn = __sinf(ang);
            float o = (d < 64) ? (val * cs - oth * sn) : (val * cs + oth * sn);
            kr[(((size_t)b * KV_ + kv) * T_ + t) * D_ + d] = f2bf(o);
        } else {                                            // V passthrough
            int idx = c - 2560;
            int kv = idx >> 7, d = idx & 127;
            vr[(((size_t)b * KV_ + kv) * T_ + t) * D_ + d] = row[c];
        }
    }
}

// ---------- flash attention (causal, GQA) ----------
__global__ __launch_bounds__(256) void attn_fa(const unsigned short* __restrict__ qr,
                                               const unsigned short* __restrict__ kr,
                                               const unsigned short* __restrict__ vr,
                                               unsigned short* __restrict__ ao) {
    __shared__ unsigned short Ks[64 * 136];    // [key][d] stride 136
    __shared__ unsigned short Vt[128 * 72];    // [d][key] stride 72 (transposed)
    __shared__ unsigned short Ps[8 * 16 * 72]; // per-wave P scratch [16 q][64 key]

    const int bh = blockIdx.x;
    const int b = bh / H_, hq = bh % H_, kv = hq / G_;
    const int i0 = blockIdx.y * 128;

    const int tid = threadIdx.x;
    const int wid = tid >> 5, lane = tid & 31;
    const int h = lane >> 4, ln = lane & 15;

    // Q fragments straight from global (row-major, lane = q row)
    const unsigned short* qbase =
        qr + (((size_t)b * H_ + hq) * T_ + i0 + wid * 16 + ln) * D_;
    v16bf qf[4];
    #pragma unroll
    for (int kk = 0; kk < 4; kk++)
        qf[kk] = ldfragA(qbase + kk * 32 + h * 16, qbase + kk * 32 + h * 16 + 8);

    v8f oacc[8];
    #pragma unroll
    for (int dn = 0; dn < 8; dn++) oacc[dn] = (v8f){0.f,0.f,0.f,0.f,0.f,0.f,0.f,0.f};
    float mrow[8], lrow[8];
    #pragma unroll
    for (int r = 0; r < 8; r++) { mrow[r] = -1e30f; lrow[r] = 0.f; }

    const float scale = 0.08838834764831845f;   // 1/sqrt(128)
    const int ntiles = i0 / 64 + 2;             // causal: keys <= i0+127

    for (int jt = 0; jt < ntiles; ++jt) {
        const int j0 = jt * 64;
        // K tile 64x128 -> LDS row-major via async global->LDS copies
        {
            int row = tid >> 2, d0 = (tid & 3) * 32;
            const unsigned short* gsrc =
                kr + (((size_t)b * KV_ + kv) * T_ + j0 + row) * D_ + d0;
            unsigned la = (unsigned)(size_t)&Ks[row * 136 + d0];
            #pragma unroll
            for (int j = 0; j < 4; j++)
                async_ld_b128(la + j * 16, (unsigned long long)(size_t)gsrc + j * 16);
        }
        // V tile 64x128 -> LDS transposed [d][key] (register path: transpose)
        {
            int key = tid >> 2, d0 = (tid & 3) * 32;
            const uint4* s = (const uint4*)(vr + (((size_t)b * KV_ + kv) * T_ + j0 + key) * D_ + d0);
            uint4 tmp[4];
            #pragma unroll
            for (int j = 0; j < 4; j++) tmp[j] = s[j];
            const unsigned short* w = (const unsigned short*)tmp;
            #pragma unroll
            for (int j = 0; j < 32; j++) Vt[(d0 + j) * 72 + key] = w[j];
        }
        wait_asynccnt0();
        __syncthreads();

        // S = Q K^T : 4 key sub-tiles x 4 D-chunks
        v8f sacc[4];
        #pragma unroll
        for (int nt = 0; nt < 4; nt++) {
            v8f s = (v8f){0.f,0.f,0.f,0.f,0.f,0.f,0.f,0.f};
            #pragma unroll
            for (int kk = 0; kk < 4; kk++) {
                v16bf bfr = *(const v16bf*)&Ks[(nt * 16 + ln) * 136 + kk * 32 + h * 16];
                s = wmma_bf16(qf[kk], bfr, s);
            }
            sacc[nt] = s;
        }

        // online softmax (rows r+8h per lane; reduce across 16 lanes)
        float alpha[8];
        #pragma unroll
        for (int r = 0; r < 8; r++) {
            int q = i0 + wid * 16 + r + 8 * h;
            float mx = -1e30f;
            #pragma unroll
            for (int nt = 0; nt < 4; nt++) {
                float sv = sacc[nt][r] * scale;
                int key = j0 + nt * 16 + ln;
                if (key > q) sv = -1e30f;
                sacc[nt][r] = sv;
                mx = fmaxf(mx, sv);
            }
            #pragma unroll
            for (int m = 1; m < 16; m <<= 1) mx = fmaxf(mx, __shfl_xor(mx, m, 32));
            float mnew = fmaxf(mrow[r], mx);
            float a = __expf(mrow[r] - mnew);
            float rs = 0.f;
            #pragma unroll
            for (int nt = 0; nt < 4; nt++) {
                float p = __expf(sacc[nt][r] - mnew);
                sacc[nt][r] = p;
                rs += p;
            }
            #pragma unroll
            for (int m = 1; m < 16; m <<= 1) rs += __shfl_xor(rs, m, 32);
            lrow[r] = lrow[r] * a + rs;
            mrow[r] = mnew;
            alpha[r] = a;
        }
        #pragma unroll
        for (int dn = 0; dn < 8; dn++)
            #pragma unroll
            for (int r = 0; r < 8; r++) oacc[dn][r] *= alpha[r];

        // P -> LDS (C layout -> row-major), then P @ V
        unsigned short* pb = &Ps[wid * 16 * 72];
        #pragma unroll
        for (int nt = 0; nt < 4; nt++)
            #pragma unroll
            for (int r = 0; r < 8; r++)
                pb[(r + 8 * h) * 72 + nt * 16 + ln] = f2bf(sacc[nt][r]);

        #pragma unroll
        for (int kc = 0; kc < 2; kc++) {
            v16bf pf = ldfragA(&pb[ln * 72 + kc * 32 + h * 16],
                               &pb[ln * 72 + kc * 32 + h * 16 + 8]);
            #pragma unroll
            for (int dn = 0; dn < 8; dn++) {
                v16bf vf = *(const v16bf*)&Vt[(dn * 16 + ln) * 72 + kc * 32 + h * 16];
                oacc[dn] = wmma_bf16(pf, vf, oacc[dn]);
            }
        }
        __syncthreads();
    }

    // O /= l ; write [B*T, H*D] bf16
    #pragma unroll
    for (int r = 0; r < 8; r++) {
        float inv = 1.0f / lrow[r];
        int t = i0 + wid * 16 + r + 8 * h;
        #pragma unroll
        for (int dn = 0; dn < 8; dn++)
            ao[((size_t)b * T_ + t) * (H_ * D_) + hq * D_ + dn * 16 + ln] =
                f2bf(oacc[dn][r] * inv);
    }
}

// ---------- host ----------
extern "C" void kernel_launch(void* const* d_in, const int* in_sizes, int n_in,
                              void* d_out, int out_size, void* d_ws, size_t ws_size,
                              hipStream_t stream) {
    const float* x  = (const float*)d_in[0];
    const float* Wq = (const float*)d_in[1];
    const float* Wk = (const float*)d_in[2];
    const float* Wv = (const float*)d_in[3];
    const float* Wo = (const float*)d_in[4];
    float* out = (float*)d_out;

    char* ws = (char*)d_ws;
    size_t off = 0;
    auto alloc = [&](size_t bytes) -> void* {
        void* p = ws + off;
        off += (bytes + 255) & ~(size_t)255;
        return p;
    };

    const size_t M = (size_t)B_ * T_;                    // 8192
    unsigned short* x16    = (unsigned short*)alloc(M * C_ * 2);
    unsigned short* wqkv16 = (unsigned short*)alloc((size_t)C_ * 3072 * 2);
    unsigned short* wo16   = (unsigned short*)alloc((size_t)C_ * C_ * 2);
    unsigned short* qkv16  = (unsigned short*)alloc(M * 3072 * 2);
    unsigned short* q_r    = (unsigned short*)alloc((size_t)B_ * H_ * T_ * D_ * 2);
    unsigned short* k_r    = (unsigned short*)alloc((size_t)B_ * KV_ * T_ * D_ * 2);
    unsigned short* v_r    = (unsigned short*)alloc((size_t)B_ * KV_ * T_ * D_ * 2);
    unsigned short* ao16   = (unsigned short*)alloc(M * C_ * 2);

    // stage 0: conversions
    {
        int n = (int)(M * C_);
        cvt_f32_bf16<<<(n + 255) / 256, 256, 0, stream>>>(x, x16, n);
    }
    pack_wqkv<<<(C_ * 3072 + 255) / 256, 256, 0, stream>>>(Wq, Wk, Wv, wqkv16);
    {
        int n = C_ * C_;
        cvt_f32_bf16<<<(n + 255) / 256, 256, 0, stream>>>(Wo, wo16, n);
    }

    // stage 1: fused QKV projection
    gemm_bf16<true><<<dim3(3072 / 128, (int)(M / 128)), 256, 0, stream>>>(
        x16, wqkv16, qkv16, (int)M, 3072, C_);

    // stage 2: RoPE + rearrange
    rope_rearrange<<<(int)M, 256, 0, stream>>>(qkv16, q_r, k_r, v_r);

    // stage 3: causal GQA flash attention
    attn_fa<<<dim3(B_ * H_, T_ / 128), 256, 0, stream>>>(q_r, k_r, v_r, ao16);

    // stage 4: output projection -> fp32
    gemm_bf16<false><<<dim3(C_ / 128, (int)(M / 128)), 256, 0, stream>>>(
        ao16, wo16, out, (int)M, C_, C_);
}
